// VectorQuantizer_36017595744268
// MI455X (gfx1250) — matrix operand order repaired and checked by reading, not compiled, not run on hardware
//
#include <hip/hip_runtime.h>
#include <hip/hip_bf16.h>
#include <stdint.h>

#define NUM_EMB 4096
#define EMB_DIM 256
#define N_IN    32768
#define LDSROW  528   // 512B row + 16B pad -> lane n hits banks 4n..4n+3 (conflict-free)

typedef __attribute__((ext_vector_type(16))) __bf16       v16bf;
typedef __attribute__((ext_vector_type(8)))  float        v8f;
typedef __attribute__((ext_vector_type(4)))  unsigned int u32x4;
typedef __attribute__((ext_vector_type(8)))  int          i32x8;
typedef __attribute__((ext_vector_type(4)))  int          i32x4;

#if defined(__has_builtin)
#if __has_builtin(__builtin_amdgcn_tensor_load_to_lds) && __has_builtin(__builtin_amdgcn_s_wait_tensorcnt)
#define VQ_USE_TDM 1
#endif
#endif
#ifndef VQ_USE_TDM
#define VQ_USE_TDM 0
#endif

// ---------------------------------------------------------------- prep ------
// Convert embedding rows to bf16, compute ||e_k||^2, zero the loss accumulator.
__global__ void vq_prep(const float* __restrict__ E, __bf16* __restrict__ Ebf,
                        float* __restrict__ enorm, float* __restrict__ loss_acc) {
  const int wid = threadIdx.x >> 5, lane = threadIdx.x & 31;
  const int row = blockIdx.x * 8 + wid;
  const float* er = E + (size_t)row * EMB_DIM;
  __bf16* br = Ebf + (size_t)row * EMB_DIM;
  float s = 0.f;
#pragma unroll
  for (int t = 0; t < 8; ++t) {
    const int d = lane + t * 32;
    const float v = er[d];
    br[d] = (__bf16)v;
    s += v * v;
  }
#pragma unroll
  for (int m = 16; m >= 1; m >>= 1) s += __shfl_xor(s, m, 32);
  if (lane == 0) enorm[row] = s;
  if (blockIdx.x == 0 && threadIdx.x == 0) *loss_acc = 0.f;
}

// ------------------------------------------------------------- TDM issue ----
#if VQ_USE_TDM
__device__ __forceinline__ void tdm_issue(const __bf16* Ebf, int kt, const void* ldsdst) {
  const uint64_t ga = (uint64_t)(uintptr_t)(Ebf + (size_t)kt * 16 * EMB_DIM);
  const unsigned int la = (unsigned int)(uintptr_t)ldsdst;  // low 32 bits = LDS offset
  u32x4 g0;
  g0.x = 1u;                                                   // count=1, no gather
  g0.y = la;                                                   // lds_addr
  g0.z = (unsigned int)ga;                                     // global_addr[31:0]
  g0.w = (unsigned int)((ga >> 32) & 0x1FFFFFFull) | (2u << 30); // addr[56:32] | type=2
  i32x8 g1;
  g1[0] = (1 << 16)        // data_size = 1 (2-byte elements)
        | (1 << 20)        // pad_enable
        | (6 << 22)        // pad_interval: every 128 dwords (512B = one tile row)
        | (3 << 25);       // pad_amount: 4 dwords (16B)
  g1[1] = EMB_DIM << 16;   // tensor_dim0 (low 16)
  g1[2] = 16 << 16;        // tensor_dim1 (low 16)
  g1[3] = EMB_DIM << 16;   // tile_dim0
  g1[4] = 16;              // tile_dim1 (tile_dim2 = 0)
  g1[5] = EMB_DIM;         // tensor_dim0_stride[31:0]
  g1[6] = 0;
  g1[7] = 0;
  const i32x4 z4 = {0, 0, 0, 0};
  const i32x8 z8 = {0, 0, 0, 0, 0, 0, 0, 0};
  // 6-arg form (clang-23 / therock-10.0 headers): groups 0..3, spare group, cpol
  __builtin_amdgcn_tensor_load_to_lds(g0, g1, z4, z4, z8, 0);
}
#endif

// --------------------------------------------------------------- argmin -----
// 8 waves x 16 rows = 128 rows per WG. Sweep all 4096 codes in 16-col tiles.
__global__ __launch_bounds__(256) void vq_argmin(
    const float* __restrict__ X, const __bf16* __restrict__ Ebf,
    const float* __restrict__ enorm, float* __restrict__ out_idx_f,
    int* __restrict__ idx_ws) {
  __shared__ __align__(16) unsigned char ebuf[2][16 * LDSROW];
  __shared__ float s_enorm[NUM_EMB];

  const int tid = threadIdx.x;
  const int wid = tid >> 5, lane = tid & 31;
  const int ln = lane & 15, hf = lane >> 4;
  const int base_m = blockIdx.x * 128 + wid * 16;

  for (int i = tid; i < NUM_EMB; i += 256) s_enorm[i] = enorm[i];

  // A fragments: 16x32 bf16 layout -> lane holds K in {c..c+7, c+16..c+23}, c = hf*8
  const float* xrow = X + (size_t)(base_m + ln) * EMB_DIM;
  const int c = hf * 8;
  v16bf afr[8];
#pragma unroll
  for (int t = 0; t < 8; ++t) {
    const float* p = xrow + t * 32 + c;
    v16bf a;
#pragma unroll
    for (int i = 0; i < 8; ++i) a[i] = (__bf16)p[i];
#pragma unroll
    for (int i = 0; i < 8; ++i) a[8 + i] = (__bf16)p[16 + i];
    afr[t] = a;
  }

  float minval[8];
  int   minidx[8];
#pragma unroll
  for (int j = 0; j < 8; ++j) { minval[j] = 3.402823466e38f; minidx[j] = 0; }

  const int KT = NUM_EMB / 16;
  // B fragment base: lane ln holds E-row (k0+ln), d-chunk offset hf*8 elements
  const unsigned char* bcol0 = &ebuf[0][0] + (size_t)ln * LDSROW + (size_t)hf * 16;
  const unsigned char* bcol1 = &ebuf[1][0] + (size_t)ln * LDSROW + (size_t)hf * 16;

#if VQ_USE_TDM
  if (wid == 0) tdm_issue(Ebf, 0, &ebuf[0][0]);
#endif

  for (int kt = 0; kt < KT; ++kt) {
#if VQ_USE_TDM
    if (wid == 0) __builtin_amdgcn_s_wait_tensorcnt(0);
    __syncthreads();
    if (wid == 0 && kt + 1 < KT) tdm_issue(Ebf, kt + 1, &ebuf[(kt + 1) & 1][0]);
    const unsigned char* bp = (kt & 1) ? bcol1 : bcol0;
#else
    __syncthreads();
    {  // cooperative padded copy of one 16x256 bf16 tile
      const uint4* src = (const uint4*)(Ebf + (size_t)kt * 16 * EMB_DIM);
      for (int i = tid; i < 512; i += 256) {
        const int r = i >> 5, ch = i & 31;
        *(uint4*)(&ebuf[0][0] + (size_t)r * LDSROW + (size_t)ch * 16) = src[i];
      }
    }
    __syncthreads();
    const unsigned char* bp = bcol0;
#endif

    v8f acc = {};
#pragma unroll
    for (int t = 0; t < 8; ++t) {
      v16bf b;
      *(uint4*)&b       = *(const uint4*)(bp + t * 64);
      *((uint4*)&b + 1) = *(const uint4*)(bp + t * 64 + 32);
      acc = __builtin_amdgcn_wmma_f32_16x16x32_bf16(false, afr[t], false, b,
                                                    (short)0, acc, false, false);
    }

    const int kcol = kt * 16 + ln;
    const float en = s_enorm[kcol];
#pragma unroll
    for (int j = 0; j < 8; ++j) {
      const float score = en - 2.0f * acc[j];
      const bool lt = score < minval[j];
      minidx[j] = lt ? kcol  : minidx[j];
      minval[j] = lt ? score : minval[j];
    }
  }

  // cross-lane argmin within each 16-lane half (ties -> smaller index)
#pragma unroll
  for (int j = 0; j < 8; ++j) {
    float v = minval[j];
    int ix = minidx[j];
#pragma unroll
    for (int m = 8; m >= 1; m >>= 1) {
      const float ov = __shfl_xor(v, m, 32);
      const int oi = __shfl_xor(ix, m, 32);
      if (ov < v || (ov == v && oi < ix)) { v = ov; ix = oi; }
    }
    if (ln == 0) {
      const int row = base_m + hf * 8 + j;
      idx_ws[row] = ix;
      out_idx_f[row] = (float)ix;
    }
  }
}

// --------------------------------------------------------------- gather -----
__global__ void vq_gather(const float* __restrict__ X, const float* __restrict__ E,
                          const int* __restrict__ idx, float* __restrict__ out_q,
                          float* __restrict__ out_in, float* __restrict__ loss_acc) {
  __shared__ float red[8];
  const int wid = threadIdx.x >> 5, lane = threadIdx.x & 31;
  const int row = blockIdx.x * 8 + wid;
  const int k = idx[row];
  const float* er = E + (size_t)k * EMB_DIM;
  const float* xr = X + (size_t)row * EMB_DIM;
  float s = 0.f;
#pragma unroll
  for (int t = 0; t < 8; ++t) {
    const int d = lane + t * 32;
    const float q = er[d], x = xr[d];
    out_q[(size_t)row * EMB_DIM + d] = q;
    out_in[(size_t)row * EMB_DIM + d] = x;
    const float df = q - x;
    s += df * df;
  }
#pragma unroll
  for (int m = 16; m >= 1; m >>= 1) s += __shfl_xor(s, m, 32);
  if (lane == 0) red[wid] = s;
  __syncthreads();
  if (threadIdx.x == 0) {
    float t = 0.f;
#pragma unroll
    for (int i = 0; i < 8; ++i) t += red[i];
    atomicAdd(loss_acc, t);
  }
}

__global__ void vq_finalize(const float* __restrict__ loss_acc, float* __restrict__ out_loss) {
  *out_loss = 1.25f * (*loss_acc) / (float)(N_IN * EMB_DIM);
}

// ---------------------------------------------------------------- launch ----
extern "C" void kernel_launch(void* const* d_in, const int* in_sizes, int n_in,
                              void* d_out, int out_size, void* d_ws, size_t ws_size,
                              hipStream_t stream) {
  const float* X = (const float*)d_in[0];
  const float* E = (const float*)d_in[1];

  float* out = (float*)d_out;
  float* out_q    = out;                                   // [N, D] quantized
  float* out_loss = out + (size_t)N_IN * EMB_DIM;          // scalar loss
  float* out_idx  = out_loss + 1;                          // [N] closest_idx (as float)
  float* out_in   = out_idx + N_IN;                        // [N, D] inputs passthrough

  unsigned char* ws = (unsigned char*)d_ws;
  __bf16* Ebf   = (__bf16*)ws;                                             // 2 MB
  float* enorm  = (float*)(ws + (size_t)NUM_EMB * EMB_DIM * 2);            // 16 KB
  int* idx_ws   = (int*)((unsigned char*)enorm + (size_t)NUM_EMB * 4);     // 128 KB
  float* loss_a = (float*)((unsigned char*)idx_ws + (size_t)N_IN * 4);     // 4 B

  vq_prep   <<<NUM_EMB / 8, 256, 0, stream>>>(E, Ebf, enorm, loss_a);
  vq_argmin <<<N_IN / 128,  256, 0, stream>>>(X, Ebf, enorm, out_idx, idx_ws);
  vq_gather <<<N_IN / 8,    256, 0, stream>>>(X, E, idx_ws, out_q, out_in, loss_a);
  vq_finalize<<<1, 1, 0, stream>>>(loss_a, out_loss);
}
